// CaptionModel_32719060861535
// MI455X (gfx1250) — matrix operand order repaired and checked by reading, not compiled, not run on hardware
//
#include <hip/hip_runtime.h>
#include <hip/hip_bf16.h>
#include <math.h>

#define BEAM 16
#define TSEQ 32
#define VSZ  50257
#define VP   50272          // V padded to multiple of 16 (3142 tiles)
#define DSZ  1024
#define NTILE_V (VP/16)     // 3142
#define TRI_PEN (-1.386f)   // -0.693 * 2.0

typedef __attribute__((ext_vector_type(16))) __bf16 v16bf;
typedef __attribute__((ext_vector_type(8)))  float  v8f;

union Frag {
  uint4 u[2];   // 32 bytes = 16 bf16
  v16bf v;
};

// ---------------------------------------------------------------------------
// Prep: transpose + fp32 -> bf16 convert.  src is [R][C] row-major fp32,
// dst is [Cp][R] bf16 (rows c >= C zero-filled).
// ---------------------------------------------------------------------------
__global__ void k_transpose_cvt(const float* __restrict__ src, __bf16* __restrict__ dst,
                                int R, int C, int Cp) {
  __shared__ float tile[32][33];
  int c0 = blockIdx.x * 32, r0 = blockIdx.y * 32;
  int tx = threadIdx.x, ty = threadIdx.y;
#pragma unroll
  for (int i = 0; i < 4; i++) {
    int r = r0 + ty + i * 8, c = c0 + tx;
    float v = 0.f;
    if (r < R && c < C) v = src[(size_t)r * C + c];
    tile[ty + i * 8][tx] = v;
  }
  __syncthreads();
#pragma unroll
  for (int i = 0; i < 4; i++) {
    int cc = c0 + ty + i * 8, rr = r0 + tx;
    if (cc < Cp && rr < R) dst[(size_t)cc * R + rr] = (__bf16)tile[tx][ty + i * 8];
  }
}

// ---------------------------------------------------------------------------
// Prep: initialize state.  logits buffer holds init_logprobs (rmax=rlsum=0 so
// logp == logits at t==0), hidden bf16, sums, beam_seq.
// ---------------------------------------------------------------------------
__global__ void k_init(const float* __restrict__ init_h, const float* __restrict__ init_lp,
                       __bf16* __restrict__ h0, float* __restrict__ logits,
                       float* rmax, float* rlsum, float* sums, int* seq) {
  int idx = blockIdx.x * blockDim.x + threadIdx.x;
  if (idx < BEAM * VP) {
    int b = idx / VP, v = idx - b * VP;
    logits[idx] = (v < VSZ) ? init_lp[(size_t)b * VSZ + v] : -1e30f;
  }
  if (idx < BEAM * DSZ) h0[idx] = (__bf16)init_h[idx];
  if (idx < TSEQ * BEAM) seq[idx] = 0;
  if (idx < BEAM) { rmax[idx] = 0.f; rlsum[idx] = 0.f; sums[idx] = 0.f; }
}

// ---------------------------------------------------------------------------
// Trigram-blocking candidate lists (per beam, tiny).
// ---------------------------------------------------------------------------
__global__ void k_trigram(const int* __restrict__ seq, int* ntri, int* tril, int t) {
  int b = threadIdx.x;
  if (b >= BEAM) return;
  int n = 0;
  if (t >= 3) {
    int a  = seq[(t - 2) * BEAM + b];
    int bb = seq[(t - 1) * BEAM + b];
    for (int p = 0; p <= t - 3; p++) {
      if (seq[p * BEAM + b] == a && seq[(p + 1) * BEAM + b] == bb) {
        tril[b * TSEQ + n] = seq[(p + 2) * BEAM + b];
        n++;
      }
    }
  }
  ntri[b] = n;
}

// ---------------------------------------------------------------------------
// Partial top-16.  64 blocks = 16 beams x 4 vocab quarters.  Per-thread
// register top-16 with LDS merge.  Flat index = b*V + v (matches reference).
// ---------------------------------------------------------------------------
__global__ void __launch_bounds__(256)
k_topk_part(const float* __restrict__ logits, const float* __restrict__ rmax,
            const float* __restrict__ rlsum, const float* __restrict__ sums,
            const int* __restrict__ ntri, const int* __restrict__ tril,
            float* __restrict__ pval, int* __restrict__ pidx, int t) {
  int b = blockIdx.x >> 2, quarter = blockIdx.x & 3;
  int tid = threadIdx.x;
  __shared__ int   s_tri[TSEQ];
  __shared__ float s_sv[256 * 16];
  __shared__ int   s_si[256 * 16];
  __shared__ float redv[256];
  __shared__ int   redi[256];
  int n = ntri[b];
  if (tid < TSEQ) s_tri[tid] = tril[b * TSEQ + tid];
  float base = sums[b], rm = rmax[b], rl = rlsum[b];
  __syncthreads();
  const int chunk = (VSZ + 3) / 4;
  int vbeg = quarter * chunk;
  int vend = vbeg + chunk; if (vend > VSZ) vend = VSZ;
  float bv[16]; int bi[16];
#pragma unroll
  for (int i = 0; i < 16; i++) { bv[i] = -3.4e38f; bi[i] = 0; }
  float vmin = -3.4e38f;
  bool masked = (t == 0) && (b > 0);
  for (int v = vbeg + tid; v < vend; v += 256) {
    float lp = logits[(size_t)b * VP + v] - rm - rl;
    if (v == VSZ - 1) lp -= 1000.f;
    int cnt = 0;
    for (int i = 0; i < n; i++) cnt += (s_tri[i] == v) ? 1 : 0;
    float cand = masked ? -1e30f : (base + lp + (float)cnt * TRI_PEN);
    if (cand > vmin) {
      bool done = false;
#pragma unroll
      for (int i = 0; i < 16; i++) {
        if (!done && bv[i] == vmin) { bv[i] = cand; bi[i] = b * VSZ + v; done = true; }
      }
      vmin = bv[0];
#pragma unroll
      for (int i = 1; i < 16; i++) vmin = fminf(vmin, bv[i]);
    }
  }
#pragma unroll
  for (int i = 0; i < 16; i++) { s_sv[tid * 16 + i] = bv[i]; s_si[tid * 16 + i] = bi[i]; }
  __syncthreads();
  for (int r = 0; r < 16; r++) {
    float m = -3.4e38f; int mi = -1;
    for (int i = tid; i < 4096; i += 256) { if (s_sv[i] > m) { m = s_sv[i]; mi = i; } }
    redv[tid] = m; redi[tid] = mi;
    __syncthreads();
    for (int s = 128; s > 0; s >>= 1) {
      if (tid < s && redv[tid + s] > redv[tid]) { redv[tid] = redv[tid + s]; redi[tid] = redi[tid + s]; }
      __syncthreads();
    }
    if (tid == 0) {
      int w = redi[0];
      pval[blockIdx.x * 16 + r] = redv[0];
      pidx[blockIdx.x * 16 + r] = (w >= 0) ? s_si[w] : 0;
      if (w >= 0) s_sv[w] = -3.4e38f;
    }
    __syncthreads();
  }
}

// ---------------------------------------------------------------------------
// Final top-16 merge + all beam bookkeeping (sums, beam_seq reorder, perm).
// ---------------------------------------------------------------------------
__global__ void __launch_bounds__(256)
k_topk_final(const float* __restrict__ pval, const int* __restrict__ pidx,
             float* sums, int* seq, int* perm, int* qcur, int* ccur, int t) {
  int tid = threadIdx.x;
  __shared__ float sv[1024]; __shared__ int si[1024];
  __shared__ float redv[256]; __shared__ int redi[256];
  __shared__ int oq[16], oc[16]; __shared__ float ov[16];
  __shared__ int stage[TSEQ * BEAM];
  for (int i = tid; i < 1024; i += 256) { sv[i] = pval[i]; si[i] = pidx[i]; }
  __syncthreads();
  for (int r = 0; r < 16; r++) {
    float m = -3.4e38f; int mi = -1;
    for (int i = tid; i < 1024; i += 256) { if (sv[i] > m) { m = sv[i]; mi = i; } }
    redv[tid] = m; redi[tid] = mi;
    __syncthreads();
    for (int s = 128; s > 0; s >>= 1) {
      if (tid < s && redv[tid + s] > redv[tid]) { redv[tid] = redv[tid + s]; redi[tid] = redi[tid + s]; }
      __syncthreads();
    }
    if (tid == 0) {
      int w = redi[0]; int f = si[w];
      ov[r] = redv[0]; oq[r] = f / VSZ; oc[r] = f % VSZ;
      sv[w] = -3.4e38f;
    }
    __syncthreads();
  }
  // physical reorder of beam_seq history rows [0, t)
  for (int i = tid; i < t * BEAM; i += 256) stage[i] = seq[i];
  __syncthreads();
  for (int i = tid; i < t * BEAM; i += 256) {
    int p = i >> 4, j = i & 15;
    seq[i] = stage[p * BEAM + oq[j]];
  }
  if (tid < BEAM) {
    seq[t * BEAM + tid] = oc[tid];
    perm[t * BEAM + tid] = oq[tid];
    qcur[tid] = oq[tid];
    ccur[tid] = oc[tid];
    sums[tid] = (oc[tid] == 0) ? -1000.f : ov[tid];
  }
}

// ---------------------------------------------------------------------------
// Store selected unaugmented logprob rows for step t (lazy beam_lp).
// ---------------------------------------------------------------------------
__global__ void __launch_bounds__(256)
k_store_lp(const float* __restrict__ logits, const float* __restrict__ rmax,
           const float* __restrict__ rlsum, const int* __restrict__ qcur,
           float* __restrict__ stored, int t) {
  long idx = (long)blockIdx.x * 256 + threadIdx.x;
  if (idx >= (long)BEAM * VSZ) return;
  int b = (int)(idx / VSZ);
  int v = (int)(idx - (long)b * VSZ);
  int q = qcur[b];
  float lp = logits[(size_t)q * VP + v] - rmax[q] - rlsum[q];
  if (v == VSZ - 1) lp -= 1000.f;
  stored[(size_t)t * BEAM * VSZ + idx] = lp;
}

// ---------------------------------------------------------------------------
// RNN hidden update: hnext = tanh(emb[c] + hcur[q] @ W_h), bf16 WMMA.
// One wave per 16-column tile of D (64 blocks x 32 threads).
// ---------------------------------------------------------------------------
__global__ void __launch_bounds__(32)
k_rnn(const __bf16* __restrict__ hcur, const __bf16* __restrict__ whT,
      const float* __restrict__ emb, const int* __restrict__ qcur,
      const int* __restrict__ ccur, __bf16* __restrict__ hnext) {
  int lane = threadIdx.x;
  int nt = blockIdx.x;
  int m = lane & 15;
  int half = lane >> 4;
  const __bf16* abase = hcur + (size_t)qcur[m] * DSZ;       // gathered source beam row
  const __bf16* bbase = whT + (size_t)(nt * 16 + m) * DSZ;  // W_h^T row = output column
  v8f acc = {0.f, 0.f, 0.f, 0.f, 0.f, 0.f, 0.f, 0.f};
  for (int k0 = 0; k0 < DSZ; k0 += 32) {
    Frag a, b;
    a.u[0] = *(const uint4*)(abase + k0 + half * 8);        // A 16x32 bf16 lane layout
    a.u[1] = *(const uint4*)(abase + k0 + 16 + half * 8);
    b.u[0] = *(const uint4*)(bbase + k0 + half * 8);        // B 32x16 bf16 lane layout
    b.u[1] = *(const uint4*)(bbase + k0 + 16 + half * 8);
    acc = __builtin_amdgcn_wmma_f32_16x16x32_bf16(false, a.v, false, b.v, (short)0, acc, false, false);
  }
  int N = nt * 16 + m;
#pragma unroll
  for (int r = 0; r < 8; r++) {
    int M = (half ? 8 : 0) + r;
    float x = acc[r] + emb[(size_t)ccur[M] * DSZ + N];
    hnext[(size_t)M * DSZ + N] = (__bf16)tanhf(x);
  }
}

// ---------------------------------------------------------------------------
// Big GEMM: logits[16][VP] = hidden[16][1024] @ W_out (bf16, L2-resident).
// 8 waves per block, one 16-column tile per wave.
// ---------------------------------------------------------------------------
__global__ void __launch_bounds__(256)
k_logits(const __bf16* __restrict__ h, const __bf16* __restrict__ woutT,
         float* __restrict__ logits) {
  int lane = threadIdx.x & 31;
  int wave = threadIdx.x >> 5;
  int nt = blockIdx.x * 8 + wave;
  if (nt >= NTILE_V) return;
  int m = lane & 15, half = lane >> 4;
  const __bf16* abase = h + (size_t)m * DSZ;
  const __bf16* bbase = woutT + (size_t)(nt * 16 + m) * DSZ;
  v8f acc = {0.f, 0.f, 0.f, 0.f, 0.f, 0.f, 0.f, 0.f};
  for (int k0 = 0; k0 < DSZ; k0 += 32) {
    Frag a, b;
    a.u[0] = *(const uint4*)(abase + k0 + half * 8);
    a.u[1] = *(const uint4*)(abase + k0 + 16 + half * 8);
    b.u[0] = *(const uint4*)(bbase + k0 + half * 8);
    b.u[1] = *(const uint4*)(bbase + k0 + 16 + half * 8);
    acc = __builtin_amdgcn_wmma_f32_16x16x32_bf16(false, a.v, false, b.v, (short)0, acc, false, false);
  }
  int vcol = nt * 16 + m;
#pragma unroll
  for (int r = 0; r < 8; r++) {
    int M = (half ? 8 : 0) + r;
    float x = (vcol < VSZ) ? acc[r] : -1e30f;
    logits[(size_t)M * VP + vcol] = x;
  }
}

// ---------------------------------------------------------------------------
// Per-row max + log-sum-exp for log_softmax (folded into later reads).
// ---------------------------------------------------------------------------
__global__ void __launch_bounds__(256)
k_rowstats(const float* __restrict__ logits, float* rmax, float* rlsum) {
  int b = blockIdx.x, tid = threadIdx.x;
  __shared__ float red[256];
  float m = -3.4e38f;
  for (int v = tid; v < VSZ; v += 256) m = fmaxf(m, logits[(size_t)b * VP + v]);
  red[tid] = m; __syncthreads();
  for (int s = 128; s > 0; s >>= 1) { if (tid < s) red[tid] = fmaxf(red[tid], red[tid + s]); __syncthreads(); }
  m = red[0]; __syncthreads();
  float sum = 0.f;
  for (int v = tid; v < VSZ; v += 256) sum += expf(logits[(size_t)b * VP + v] - m);
  red[tid] = sum; __syncthreads();
  for (int s = 128; s > 0; s >>= 1) { if (tid < s) red[tid] += red[tid + s]; __syncthreads(); }
  if (tid == 0) { rmax[b] = m; rlsum[b] = logf(red[0]); }
}

// ---------------------------------------------------------------------------
// End: compose backpointer chains, emit beam_seq (as floats) and sums.
// ---------------------------------------------------------------------------
__global__ void k_compose(const int* __restrict__ perm, const int* __restrict__ seq,
                          const float* __restrict__ sums, int* __restrict__ g,
                          float* __restrict__ out) {
  int tid = threadIdx.x;
  if (tid < TSEQ * BEAM) out[tid] = (float)seq[tid];
  if (tid < BEAM) {
    int idx = tid;
    g[(TSEQ - 1) * BEAM + tid] = idx;
    for (int tt = TSEQ - 2; tt >= 0; tt--) {
      idx = perm[(tt + 1) * BEAM + idx];
      g[tt * BEAM + tid] = idx;
    }
    out[(size_t)TSEQ * BEAM + (size_t)TSEQ * BEAM * VSZ + tid] = sums[tid];
  }
}

// ---------------------------------------------------------------------------
// End: single gather producing final beam_lp ordering.
// ---------------------------------------------------------------------------
__global__ void __launch_bounds__(256)
k_gather(const float* __restrict__ stored, const int* __restrict__ g,
         float* __restrict__ out) {
  size_t idx = (size_t)blockIdx.x * 256 + threadIdx.x;
  const size_t total = (size_t)TSEQ * BEAM * VSZ;
  if (idx >= total) return;
  int t = (int)(idx / ((size_t)BEAM * VSZ));
  size_t rem = idx - (size_t)t * BEAM * VSZ;
  int j = (int)(rem / VSZ);
  int v = (int)(rem - (size_t)j * VSZ);
  out[(size_t)TSEQ * BEAM + idx] =
      stored[(size_t)t * BEAM * VSZ + (size_t)g[t * BEAM + j] * VSZ + v];
}

// ---------------------------------------------------------------------------
extern "C" void kernel_launch(void* const* d_in, const int* in_sizes, int n_in,
                              void* d_out, int out_size, void* d_ws, size_t ws_size,
                              hipStream_t stream) {
  const float* init_h  = (const float*)d_in[0];
  const float* init_lp = (const float*)d_in[1];
  const float* emb     = (const float*)d_in[2];
  const float* W_h     = (const float*)d_in[3];
  const float* W_out   = (const float*)d_in[4];
  float* out = (float*)d_out;

  char* p = (char*)d_ws;
  auto alloc = [&](size_t bytes) -> void* {
    void* r = (void*)p;
    p += (bytes + 255) & ~(size_t)255;
    return r;
  };
  __bf16* woutT = (__bf16*)alloc((size_t)VP * DSZ * 2);       // W_out^T bf16 (~98 MB, L2-resident)
  __bf16* whT   = (__bf16*)alloc((size_t)DSZ * DSZ * 2);      // W_h^T bf16
  __bf16* hbf0  = (__bf16*)alloc((size_t)BEAM * DSZ * 2);
  __bf16* hbf1  = (__bf16*)alloc((size_t)BEAM * DSZ * 2);
  float*  logits = (float*)alloc((size_t)BEAM * VP * 4);
  float*  rmax   = (float*)alloc(BEAM * 4);
  float*  rlsum  = (float*)alloc(BEAM * 4);
  float*  sums   = (float*)alloc(BEAM * 4);
  int*    seq    = (int*)alloc(TSEQ * BEAM * 4);
  int*    perm   = (int*)alloc(TSEQ * BEAM * 4);
  int*    qcur   = (int*)alloc(BEAM * 4);
  int*    ccur   = (int*)alloc(BEAM * 4);
  int*    ntri   = (int*)alloc(BEAM * 4);
  int*    tril   = (int*)alloc(BEAM * TSEQ * 4);
  float*  pval   = (float*)alloc(64 * 16 * 4);
  int*    pidx   = (int*)alloc(64 * 16 * 4);
  float*  stored = (float*)alloc((size_t)TSEQ * BEAM * VSZ * 4);  // ~98 MB
  int*    g      = (int*)alloc(TSEQ * BEAM * 4);
  (void)ws_size; (void)n_in; (void)in_sizes; (void)out_size;

  dim3 tpt(32, 8);
  k_transpose_cvt<<<dim3((VP + 31) / 32, (DSZ + 31) / 32), tpt, 0, stream>>>(W_out, woutT, DSZ, VSZ, VP);
  k_transpose_cvt<<<dim3((DSZ + 31) / 32, (DSZ + 31) / 32), tpt, 0, stream>>>(W_h, whT, DSZ, DSZ, DSZ);
  k_init<<<(BEAM * VP + 255) / 256, 256, 0, stream>>>(init_h, init_lp, hbf0, logits, rmax, rlsum, sums, seq);

  for (int t = 0; t < TSEQ; t++) {
    __bf16* hc = (t & 1) ? hbf1 : hbf0;
    __bf16* hn = (t & 1) ? hbf0 : hbf1;
    k_trigram<<<1, 32, 0, stream>>>(seq, ntri, tril, t);
    k_topk_part<<<64, 256, 0, stream>>>(logits, rmax, rlsum, sums, ntri, tril, pval, pidx, t);
    k_topk_final<<<1, 256, 0, stream>>>(pval, pidx, sums, seq, perm, qcur, ccur, t);
    k_store_lp<<<(BEAM * VSZ + 255) / 256, 256, 0, stream>>>(logits, rmax, rlsum, qcur, stored, t);
    if (t < TSEQ - 1) {
      k_rnn<<<DSZ / 16, 32, 0, stream>>>(hc, whT, emb, qcur, ccur, hn);
      k_logits<<<(NTILE_V + 7) / 8, 256, 0, stream>>>(hn, woutT, logits);
      k_rowstats<<<BEAM, 256, 0, stream>>>(logits, rmax, rlsum);
    }
  }

  k_compose<<<1, 512, 0, stream>>>(perm, seq, sums, g, out);
  k_gather<<<(int)(((size_t)TSEQ * BEAM * VSZ + 255) / 256), 256, 0, stream>>>(stored, g, out);
}
